// multiheadlatentattention_59803124629721
// MI455X (gfx1250) — matrix operand order repaired and checked by reading, not compiled, not run on hardware
//
#include <hip/hip_runtime.h>

// ---------------------------------------------------------------------------
// MLA forward for MI455X (gfx1250, wave32).  All matmuls use
// v_wmma_f32_16x16x32_bf16 (fp32 accumulate).  fp32 inputs are converted to
// bf16 on the LDS staging path; intermediates live in d_ws as bf16.
// ---------------------------------------------------------------------------

typedef __attribute__((ext_vector_type(16))) __bf16 bf16x16;
typedef __attribute__((ext_vector_type(8)))  float  f32x8;

union BFrag { bf16x16 v; uint4 u[2]; };
union CFrag { f32x8 v; float f[8]; };

static __device__ __forceinline__ unsigned short f2bf(float x) {
  unsigned int u = __float_as_uint(x);
  u += 0x7FFFu + ((u >> 16) & 1u);          // round-to-nearest-even
  return (unsigned short)(u >> 16);
}

static __device__ __forceinline__ f32x8 wmma_bf16(bf16x16 a, bf16x16 b, f32x8 c) {
  // 8 args: (neg_a, A, neg_b, B, c_mod, C, reuse_a, reuse_b)
  return __builtin_amdgcn_wmma_f32_16x16x32_bf16(false, a, false, b, (short)0, c,
                                                 false, false);
}

// -- load 8 contiguous elements as packed bf16 (uint4) ----------------------
static __device__ __forceinline__ uint4 load8_bf(const unsigned short* p) {
  return *reinterpret_cast<const uint4*>(p);
}
static __device__ __forceinline__ uint4 load8_bf(const float* p) {
  const float4* q = reinterpret_cast<const float4*>(p);
  float4 f0 = q[0], f1 = q[1];
  uint4 r;
  r.x = (unsigned)f2bf(f0.x) | ((unsigned)f2bf(f0.y) << 16);
  r.y = (unsigned)f2bf(f0.z) | ((unsigned)f2bf(f0.w) << 16);
  r.z = (unsigned)f2bf(f1.x) | ((unsigned)f2bf(f1.y) << 16);
  r.w = (unsigned)f2bf(f1.z) | ((unsigned)f2bf(f1.w) << 16);
  return r;
}

// -- load 16 contiguous elements into 16 bf16 shorts ------------------------
static __device__ __forceinline__ void load16_bf(const unsigned short* p,
                                                 unsigned short* o) {
  uint4 a = reinterpret_cast<const uint4*>(p)[0];
  uint4 b = reinterpret_cast<const uint4*>(p)[1];
  o[0]=(unsigned short)a.x; o[1]=(unsigned short)(a.x>>16);
  o[2]=(unsigned short)a.y; o[3]=(unsigned short)(a.y>>16);
  o[4]=(unsigned short)a.z; o[5]=(unsigned short)(a.z>>16);
  o[6]=(unsigned short)a.w; o[7]=(unsigned short)(a.w>>16);
  o[8]=(unsigned short)b.x; o[9]=(unsigned short)(b.x>>16);
  o[10]=(unsigned short)b.y; o[11]=(unsigned short)(b.y>>16);
  o[12]=(unsigned short)b.z; o[13]=(unsigned short)(b.z>>16);
  o[14]=(unsigned short)b.w; o[15]=(unsigned short)(b.w>>16);
}
static __device__ __forceinline__ void load16_bf(const float* p,
                                                 unsigned short* o) {
  const float4* q = reinterpret_cast<const float4*>(p);
#pragma unroll
  for (int v = 0; v < 4; ++v) {
    float4 f = q[v];
    o[v*4+0] = f2bf(f.x); o[v*4+1] = f2bf(f.y);
    o[v*4+2] = f2bf(f.z); o[v*4+3] = f2bf(f.w);
  }
}

static __device__ __forceinline__ void store_out(float* p, float v) { *p = v; }
static __device__ __forceinline__ void store_out(unsigned short* p, float v) {
  *p = f2bf(v);
}

// ---------------------------------------------------------------------------
// GEMM:  C[M,N] = A[M,K] @ B[K,N] + bias[N]
// BM=64 BN=128 BK=32.  8 waves, each owns a 32x32 tile (2x2 WMMA frags).
// LDS rows padded to 40 elements (80 B, 16B-aligned, conflict-mitigating).
// ---------------------------------------------------------------------------
template <typename TA, typename TB, typename TO>
__global__ __launch_bounds__(256)
void gemm_bias_wmma(const TA* __restrict__ A, const TB* __restrict__ B,
                    const float* __restrict__ bias, TO* __restrict__ C,
                    int M, int N, int K) {
  constexpr int LDA = 40, LDB = 40;
  __shared__ unsigned short sA[64 * LDA];
  __shared__ unsigned short sB[128 * LDB];

  const int tid  = threadIdx.x;
  const int lane = tid & 31;
  const int wave = tid >> 5;
  const int wm = (wave & 1) * 32;       // wave row tile within block
  const int wn = (wave >> 1) * 32;      // wave col tile within block
  const int row0 = blockIdx.y * 64;
  const int n0   = blockIdx.x * 128;
  const int half = lane >> 4, l15 = lane & 15;

  CFrag acc[2][2];
#pragma unroll
  for (int i = 0; i < 2; ++i)
#pragma unroll
    for (int j = 0; j < 2; ++j)
#pragma unroll
      for (int r = 0; r < 8; ++r) acc[i][j].f[r] = 0.0f;

  for (int kc = 0; kc < K; kc += 32) {
    // ---- stage A tile 64x32 (8 elems/thread, one b128 store) ----
    {
      int idx = tid * 8, r = idx >> 5, c = idx & 31;
      uint4 d = load8_bf(A + (size_t)(row0 + r) * K + kc + c);
      *reinterpret_cast<uint4*>(&sA[r * LDA + c]) = d;
    }
    // ---- stage B tile 32x128 transposed -> sB[n][k] (16 elems/thread) ----
    {
      int idx = tid * 16, r = idx >> 7, c = idx & 127;
      unsigned short t[16];
      load16_bf(B + (size_t)(kc + r) * N + n0 + c, t);
#pragma unroll
      for (int i = 0; i < 16; ++i) sB[(c + i) * LDB + r] = t[i];
    }
    __syncthreads();

    BFrag af[2], bf[2];
#pragma unroll
    for (int i = 0; i < 2; ++i) {  // A layout: K = half*8+0..7, 16+half*8+0..7
      int m = wm + i * 16 + l15;
      af[i].u[0] = *reinterpret_cast<uint4*>(&sA[m * LDA + half * 8]);
      af[i].u[1] = *reinterpret_cast<uint4*>(&sA[m * LDA + 16 + half * 8]);
    }
#pragma unroll
    for (int j = 0; j < 2; ++j) {  // B layout: K = half*8+0..7, 16+half*8+0..7
      int n = wn + j * 16 + l15;
      bf[j].u[0] = *reinterpret_cast<uint4*>(&sB[n * LDB + half * 8]);
      bf[j].u[1] = *reinterpret_cast<uint4*>(&sB[n * LDB + 16 + half * 8]);
    }
#pragma unroll
    for (int i = 0; i < 2; ++i)
#pragma unroll
      for (int j = 0; j < 2; ++j)
        acc[i][j].v = wmma_bf16(af[i].v, bf[j].v, acc[i][j].v);
    __syncthreads();
  }

  // ---- epilogue: + bias, store (C layout: m = half*8 + r, n = l15) ----
#pragma unroll
  for (int j = 0; j < 2; ++j) {
    int ng = n0 + wn + j * 16 + l15;
    float bj = bias[ng];
#pragma unroll
    for (int i = 0; i < 2; ++i)
#pragma unroll
      for (int r = 0; r < 8; ++r) {
        int mg = row0 + wm + i * 16 + half * 8 + r;
        store_out(C + (size_t)mg * N + ng, acc[i][j].f[r] + bj);
      }
  }
}

// ---------------------------------------------------------------------------
// Causal flash attention over latent heads: hd = 64, 16 heads, dl = 1024.
// Block = 8 waves; each wave owns 16 query rows (128 rows / block).
// Streams 32-key chunks: S = Q K^T (2 WMMA col-tiles x 2 k-steps),
// online softmax in C-layout, P re-shaped via wave-private LDS,
// O += P V (4 WMMA col-tiles over hd).
// ---------------------------------------------------------------------------
__global__ __launch_bounds__(256)
void mla_flash_attn(const unsigned short* __restrict__ Q,
                    const unsigned short* __restrict__ Km,
                    const unsigned short* __restrict__ V,
                    unsigned short* __restrict__ O) {
  constexpr int HD = 64, DLT = 1024, SEQL = 2048;
  constexpr int LDK = 72, LDV = 40, LDP = 40;
  __shared__ unsigned short sK[32 * LDK];       // K chunk [s][hd]
  __shared__ unsigned short sVT[64 * LDV];      // V chunk transposed [hd][s]
  __shared__ unsigned short sP[8 * 16 * LDP];   // per-wave P staging

  const int tid = threadIdx.x, lane = tid & 31, wave = tid >> 5;
  const int half = lane >> 4, l15 = lane & 15;
  const int qt = blockIdx.x, h = blockIdx.y, b = blockIdx.z;
  const size_t base = (size_t)b * SEQL * DLT + (size_t)h * HD;
  const int qbase = qt * 128 + wave * 16;

  // Q fragments for the two hd k-steps (A layout)
  BFrag aQ[2];
  {
    const unsigned short* qrow = Q + base + (size_t)(qbase + l15) * DLT;
#pragma unroll
    for (int t = 0; t < 2; ++t) {
      aQ[t].u[0] = *reinterpret_cast<const uint4*>(qrow + t * 32 + half * 8);
      aQ[t].u[1] = *reinterpret_cast<const uint4*>(qrow + t * 32 + 16 + half * 8);
    }
  }

  CFrag acc[4];
#pragma unroll
  for (int j = 0; j < 4; ++j)
#pragma unroll
    for (int r = 0; r < 8; ++r) acc[j].f[r] = 0.0f;
  float rmax[8], rsum[8];
#pragma unroll
  for (int r = 0; r < 8; ++r) { rmax[r] = -__builtin_inff(); rsum[r] = 0.0f; }

  const int send = (qt + 1) * 128;
  for (int s0 = 0; s0 < send; s0 += 32) {
    __syncthreads();
    {  // stage K chunk 32x64 (8 elems/thread)
      int idx = tid * 8, r = idx >> 6, c = idx & 63;
      *reinterpret_cast<uint4*>(&sK[r * LDK + c]) =
          *reinterpret_cast<const uint4*>(Km + base + (size_t)(s0 + r) * DLT + c);
    }
    {  // stage V chunk transposed -> sVT[hd][s]
      int idx = tid * 8, r = idx >> 6, c = idx & 63;
      uint4 d = *reinterpret_cast<const uint4*>(V + base + (size_t)(s0 + r) * DLT + c);
      unsigned short e[8];
      e[0]=(unsigned short)d.x; e[1]=(unsigned short)(d.x>>16);
      e[2]=(unsigned short)d.y; e[3]=(unsigned short)(d.y>>16);
      e[4]=(unsigned short)d.z; e[5]=(unsigned short)(d.z>>16);
      e[6]=(unsigned short)d.w; e[7]=(unsigned short)(d.w>>16);
#pragma unroll
      for (int i = 0; i < 8; ++i) sVT[(c + i) * LDV + r] = e[i];
    }
    __syncthreads();

    // ---- S = Q K^T : two 16-wide key tiles ----
    CFrag sc[2];
#pragma unroll
    for (int j = 0; j < 2; ++j) {
#pragma unroll
      for (int r = 0; r < 8; ++r) sc[j].f[r] = 0.0f;
      int n = j * 16 + l15;
#pragma unroll
      for (int t = 0; t < 2; ++t) {
        BFrag bk;
        bk.u[0] = *reinterpret_cast<uint4*>(&sK[n * LDK + t * 32 + half * 8]);
        bk.u[1] = *reinterpret_cast<uint4*>(&sK[n * LDK + t * 32 + 16 + half * 8]);
        sc[j].v = wmma_bf16(aQ[t].v, bk.v, sc[j].v);
      }
    }

    // ---- causal mask + online softmax (per row m = half*8 + r) ----
    const float scl = 0.125f;  // 1/sqrt(64)
    float p0[8], p1[8];
#pragma unroll
    for (int r = 0; r < 8; ++r) {
      int qg = qbase + half * 8 + r;
      float v0 = ((s0 + l15)      <= qg) ? sc[0].f[r] * scl : -__builtin_inff();
      float v1 = ((s0 + 16 + l15) <= qg) ? sc[1].f[r] * scl : -__builtin_inff();
      float mx = fmaxf(v0, v1);
#pragma unroll
      for (int m = 1; m < 16; m <<= 1) mx = fmaxf(mx, __shfl_xor(mx, m, 32));
      float mnew = fmaxf(rmax[r], mx);
      float e0 = __expf(v0 - mnew), e1 = __expf(v1 - mnew);
      float corr = __expf(rmax[r] - mnew);
      float ps = e0 + e1;
#pragma unroll
      for (int m = 1; m < 16; m <<= 1) ps += __shfl_xor(ps, m, 32);
      rsum[r] = rsum[r] * corr + ps;
      rmax[r] = mnew;
      p0[r] = e0; p1[r] = e1;
#pragma unroll
      for (int j = 0; j < 4; ++j) acc[j].f[r] *= corr;
    }

    // ---- C-layout P -> A-layout via wave-private LDS (in-order per wave) --
    unsigned short* wp = &sP[wave * 16 * LDP];
#pragma unroll
    for (int r = 0; r < 8; ++r) {
      int m = half * 8 + r;
      wp[m * LDP + l15]      = f2bf(p0[r]);
      wp[m * LDP + 16 + l15] = f2bf(p1[r]);
    }
    asm volatile("s_wait_dscnt 0" ::: "memory");
    BFrag aP;
    aP.u[0] = *reinterpret_cast<uint4*>(&wp[l15 * LDP + half * 8]);
    aP.u[1] = *reinterpret_cast<uint4*>(&wp[l15 * LDP + 16 + half * 8]);

    // ---- O += P @ V over 4 hd tiles ----
#pragma unroll
    for (int j = 0; j < 4; ++j) {
      int n = j * 16 + l15;
      BFrag bv;
      bv.u[0] = *reinterpret_cast<uint4*>(&sVT[n * LDV + half * 8]);
      bv.u[1] = *reinterpret_cast<uint4*>(&sVT[n * LDV + 16 + half * 8]);
      acc[j].v = wmma_bf16(aP.v, bv.v, acc[j].v);
    }
  }

  // ---- normalize and store bf16 ----
#pragma unroll
  for (int r = 0; r < 8; ++r) {
    float inv = 1.0f / rsum[r];
    int mg = qbase + half * 8 + r;
    unsigned short* orow = O + base + (size_t)mg * DLT;
#pragma unroll
    for (int j = 0; j < 4; ++j) orow[j * 16 + l15] = f2bf(acc[j].f[r] * inv);
  }
}

// ---------------------------------------------------------------------------
extern "C" void kernel_launch(void* const* d_in, const int* in_sizes, int n_in,
                              void* d_out, int out_size, void* d_ws,
                              size_t ws_size, hipStream_t stream) {
  constexpr int D_IN = 2048, D_OUT = 2048, DL = 1024, SEQ = 2048, BATCH = 2;
  constexpr int M = BATCH * SEQ;  // 4096 rows

  const float* x   = (const float*)d_in[0];
  const float* wk  = (const float*)d_in[1];  const float* bk  = (const float*)d_in[2];
  const float* wq  = (const float*)d_in[3];  const float* bq  = (const float*)d_in[4];
  const float* wv  = (const float*)d_in[5];  const float* bv  = (const float*)d_in[6];
  const float* wkl = (const float*)d_in[7];  const float* bkl = (const float*)d_in[8];
  const float* wql = (const float*)d_in[9];  const float* bql = (const float*)d_in[10];
  const float* wvl = (const float*)d_in[11]; const float* bvl = (const float*)d_in[12];
  const float* wo  = (const float*)d_in[13]; const float* bo  = (const float*)d_in[14];
  float* out = (float*)d_out;

  // scratch layout (bf16 as unsigned short)
  unsigned short* ws = (unsigned short*)d_ws;
  unsigned short* tmp  = ws;                              // [M, 2048]
  unsigned short* qlat = tmp  + (size_t)M * D_OUT;        // [M, 1024]
  unsigned short* klat = qlat + (size_t)M * DL;
  unsigned short* vlat = klat + (size_t)M * DL;
  unsigned short* attn = vlat + (size_t)M * DL;
  (void)ws_size; (void)in_sizes; (void)n_in; (void)out_size;

  dim3 blk(256);
  dim3 gFull(D_OUT / 128, M / 64);   // (16, 64)
  dim3 gLat(DL / 128, M / 64);       // (8, 64)

  // K path
  gemm_bias_wmma<float, float, unsigned short><<<gFull, blk, 0, stream>>>(
      x, wk, bk, tmp, M, D_OUT, D_IN);
  gemm_bias_wmma<unsigned short, float, unsigned short><<<gLat, blk, 0, stream>>>(
      tmp, wkl, bkl, klat, M, DL, D_OUT);
  // Q path
  gemm_bias_wmma<float, float, unsigned short><<<gFull, blk, 0, stream>>>(
      x, wq, bq, tmp, M, D_OUT, D_IN);
  gemm_bias_wmma<unsigned short, float, unsigned short><<<gLat, blk, 0, stream>>>(
      tmp, wql, bql, qlat, M, DL, D_OUT);
  // V path
  gemm_bias_wmma<float, float, unsigned short><<<gFull, blk, 0, stream>>>(
      x, wv, bv, tmp, M, D_OUT, D_IN);
  gemm_bias_wmma<unsigned short, float, unsigned short><<<gLat, blk, 0, stream>>>(
      tmp, wvl, bvl, vlat, M, DL, D_OUT);

  // causal flash attention: grid (q-tiles, heads, batch)
  mla_flash_attn<<<dim3(SEQ / 128, 16, BATCH), blk, 0, stream>>>(qlat, klat,
                                                                 vlat, attn);

  // output projection -> fp32 d_out
  gemm_bias_wmma<unsigned short, float, float><<<gFull, blk, 0, stream>>>(
      attn, wo, bo, out, M, D_OUT, DL);
}
